// LongcatMoe_49065706390113
// MI455X (gfx1250) — matrix operand order repaired and checked by reading, not compiled, not run on hardware
//
#include <hip/hip_runtime.h>
#include <hip/hip_bf16.h>
#include <cstdint>

#define T_TOK   4096
#define H_DIM   1024
#define I_DIM   512
#define E_RTD   64
#define EZ_TOT  80
#define TOPK    2
#define CAPACITY 320
#define SCALE_F 1.5f
#define MT      4            // M-tiles (16 rows each) per block
#define MG      (MT * 16)    // 64 rows per block-group; CAP/MG = 5 exactly

#if defined(__has_builtin)
#if __has_builtin(__builtin_amdgcn_global_load_async_to_lds_b128) && \
    __has_builtin(__builtin_amdgcn_s_wait_asynccnt)
#define USE_ASYNC_LDS 1
#endif
#endif

typedef __attribute__((ext_vector_type(16))) __bf16 v16bf;
typedef __attribute__((ext_vector_type(8)))  float  v8f;
typedef __attribute__((__vector_size__(4 * sizeof(int)))) int v4i;

union BFrag { v16bf v; unsigned int u[8]; };

__device__ __forceinline__ unsigned int pack2_bf16(float lo, float hi) {
    // one v_perm_b32: result = {hi[31:16], lo[31:16]}  (truncating f32->bf16 pair)
    return __builtin_amdgcn_perm(__builtin_bit_cast(unsigned int, hi),
                                 __builtin_bit_cast(unsigned int, lo),
                                 0x07060302u);
}
__device__ __forceinline__ unsigned short to_bf16(float f) {
    return (unsigned short)(__builtin_bit_cast(unsigned int, f) >> 16);
}
__device__ __forceinline__ float fast_sigmoid(float x) {
    return __builtin_amdgcn_rcpf(1.f + __expf(-x));   // v_exp + v_add + v_rcp (no exact div)
}

// ---------------- 1. Router: logits -> sigmoid -> top-2 (biased) ----------------
__global__ __launch_bounds__(128) void router_kernel(
    const float* __restrict__ hs, const float* __restrict__ rw,
    const float* __restrict__ bias, int* __restrict__ top_idx,
    float* __restrict__ top_w, float* __restrict__ zero_coef) {
    __shared__ float row[H_DIM];
    __shared__ float s_score[EZ_TOT];
    __shared__ float s_bias[EZ_TOT];
    const int t = blockIdx.x;
    for (int i = threadIdx.x; i < H_DIM; i += 128) row[i] = hs[(size_t)t * H_DIM + i];
    __syncthreads();
    if (threadIdx.x < EZ_TOT) {
        const int e = threadIdx.x;
        const float* w = rw + (size_t)e * H_DIM;
        float acc = 0.f;
        for (int i = 0; i < H_DIM; i += 4)
            acc += row[i]*w[i] + row[i+1]*w[i+1] + row[i+2]*w[i+2] + row[i+3]*w[i+3];
        float s = fast_sigmoid(acc);
        s_score[e] = s;
        s_bias[e]  = s + bias[e];
    }
    __syncthreads();
    if (threadIdx.x == 0) {
        int i0 = 0, i1 = 0;
        float b0 = -1e30f, b1 = -1e30f;
        for (int e = 0; e < EZ_TOT; ++e) {       // strict '>' => first-max, matches top_k ties
            float b = s_bias[e];
            if (b > b0)      { b1 = b0; i1 = i0; b0 = b; i0 = e; }
            else if (b > b1) { b1 = b; i1 = e; }
        }
        float w0 = s_score[i0], w1 = s_score[i1];
        top_idx[t*2+0] = i0; top_idx[t*2+1] = i1;
        top_w[t*2+0]   = w0; top_w[t*2+1]   = w1;
        float zc = 0.f;
        if (i0 >= E_RTD) zc += w0;
        if (i1 >= E_RTD) zc += w1;
        zero_coef[t] = zc;
    }
}

// ------- 2. Deterministic dispatch: one thread/expert scans slots in token order -------
__global__ __launch_bounds__(64) void dispatch_kernel(
    const int* __restrict__ top_idx, const float* __restrict__ top_w,
    int* __restrict__ cnt, int* __restrict__ slot_tok, float* __restrict__ slot_w) {
    const int e = threadIdx.x;
    if (e >= E_RTD) return;
    int c = 0;
    for (int s = 0; s < T_TOK * TOPK; ++s) {
        if (top_idx[s] == e) {
            if (c < CAPACITY) {
                slot_tok[e*CAPACITY + c] = s >> 1;
                slot_w[e*CAPACITY + c]   = SCALE_F * top_w[s];
            }
            ++c;                                   // overflow rows dropped (like reference)
        }
    }
    cnt[e] = c;
}

// ---------------- 3. out = SCALE * zero_coef * x  (initializes d_out) ----------------
__global__ __launch_bounds__(256) void out_init_kernel(
    const float* __restrict__ hs, const float* __restrict__ zero_coef,
    float* __restrict__ out) {
    const int i = blockIdx.x * 256 + threadIdx.x;       // float4 index
    const int t = i >> 8;                               // / (H_DIM/4)
    const float c = SCALE_F * zero_coef[t];
    float4 v = ((const float4*)hs)[i];
    v.x *= c; v.y *= c; v.z *= c; v.w *= c;
    ((float4*)out)[i] = v;
}

// ---------------- 4. Gate/Up GEMM + SiLU fuse (bf16 WMMA, 4x B reuse) ----------------
// grid = (I/64, CAP/64, E), block = 128 (4 waves x 16 N-cols, 64 M-rows per block)
__global__ __launch_bounds__(128) void gateup_kernel(
    const float* __restrict__ hs, const float* __restrict__ wg, const float* __restrict__ wu,
    const int* __restrict__ cnt, const int* __restrict__ slot_tok,
    unsigned short* __restrict__ act) {
    const int e  = blockIdx.z;
    const int m0 = blockIdx.y * MG;
    int cnte = cnt[e]; if (cnte > CAPACITY) cnte = CAPACITY;
    if (m0 >= cnte) return;                         // uniform exit: EXEC stays full for WMMA

    const int wave  = threadIdx.x >> 5;
    const int lane  = threadIdx.x & 31;
    const int nBase = blockIdx.x * 64 + wave * 16;

    __shared__ unsigned short aTile[MG][32];        // 64x32 bf16 A panel (4 KB)
    const int r  = threadIdx.x >> 1;                // row 0..63, two threads per row
    const int kq = (threadIdx.x & 1) * 16;          // 16 contiguous K per thread
    const int tok = (m0 + r < cnte) ? slot_tok[e*CAPACITY + m0 + r] : -1;

    unsigned int* dstu = (unsigned int*)&aTile[r][kq];
    if (tok < 0) {                                  // zero invalid rows ONCE (they never change)
        #pragma unroll
        for (int i = 0; i < 8; ++i) dstu[i] = 0u;
    }

    const int halfSel = lane >> 4;                  // lane half selects K groups (contiguous k = 16h + s, same for A and B)
    const int mA      = lane & 15;
    const int kaBase  = halfSel * 16;               // A: K {0..15} vs {16..31}
    const int kbBase  = halfSel * 16;               // B: K {0..15} vs {16..31}
    const int nCol    = nBase + (lane & 15);
    const float* __restrict__ wgE = wg + (size_t)e * H_DIM * I_DIM + nCol;
    const float* __restrict__ wuE = wu + (size_t)e * H_DIM * I_DIM + nCol;

    v8f cg[MT]; v8f cu[MT];
    #pragma unroll
    for (int mt = 0; mt < MT; ++mt) { cg[mt] = (v8f){}; cu[mt] = (v8f){}; }

    for (int ks = 0; ks < H_DIM; ks += 32) {
        // cooperative A panel load: fp32 -> packed bf16 dword stores (valid rows only)
        if (tok >= 0) {
            const float4* src = (const float4*)(hs + (size_t)tok * H_DIM + ks + kq);
            float4 f0 = src[0], f1 = src[1], f2 = src[2], f3 = src[3];
            dstu[0] = pack2_bf16(f0.x, f0.y); dstu[1] = pack2_bf16(f0.z, f0.w);
            dstu[2] = pack2_bf16(f1.x, f1.y); dstu[3] = pack2_bf16(f1.z, f1.w);
            dstu[4] = pack2_bf16(f2.x, f2.y); dstu[5] = pack2_bf16(f2.z, f2.w);
            dstu[6] = pack2_bf16(f3.x, f3.y); dstu[7] = pack2_bf16(f3.z, f3.w);
        }
        __syncthreads();
        // B fragments: loaded ONCE per K-step, reused by 4 WMMAs each
        BFrag bg, bu;
        #pragma unroll
        for (int j = 0; j < 8; ++j) {
            const size_t koff = (size_t)(ks + kbBase + 2*j) * I_DIM;
            bg.u[j] = pack2_bf16(wgE[koff], wgE[koff + I_DIM]);
            bu.u[j] = pack2_bf16(wuE[koff], wuE[koff + I_DIM]);
        }
        #pragma unroll
        for (int mt = 0; mt < MT; ++mt) {
            BFrag af;
            const unsigned short* ar = &aTile[mt*16 + mA][0];
            #pragma unroll
            for (int j = 0; j < 4; ++j) {           // lowers to 2x ds_load_b128
                af.u[j]     = *(const unsigned int*)&ar[kaBase + 2*j];
                af.u[4 + j] = *(const unsigned int*)&ar[kaBase + 8 + 2*j];
            }
            cg[mt] = __builtin_amdgcn_wmma_f32_16x16x32_bf16(false, af.v, false, bg.v, (short)0, cg[mt], false, false);
            cu[mt] = __builtin_amdgcn_wmma_f32_16x16x32_bf16(false, af.v, false, bu.v, (short)0, cu[mt], false, false);
        }
        __syncthreads();
    }
    // epilogue: act = silu(g) * u, stored bf16; C layout: vgpr j -> M = j + 8*halfSel
    const int mTop = halfSel * 8;
    #pragma unroll
    for (int mt = 0; mt < MT; ++mt) {
        unsigned short* dst = act + (size_t)(e * CAPACITY + m0 + mt*16 + mTop) * I_DIM + nCol;
        #pragma unroll
        for (int j = 0; j < 8; ++j) {
            float g = cg[mt][j], u = cu[mt][j];
            dst[(size_t)j * I_DIM] = to_bf16(g * fast_sigmoid(g) * u);
        }
    }
}

// ---------------- 5. Down GEMM + weighted scatter-add (bf16 WMMA, 4x B reuse) ----------------
// grid = (H/64, CAP/64, E), block = 128
__global__ __launch_bounds__(128) void down_kernel(
    const unsigned short* __restrict__ act, const float* __restrict__ wd,
    const int* __restrict__ cnt, const int* __restrict__ slot_tok,
    const float* __restrict__ slot_w, float* __restrict__ out) {
    const int e  = blockIdx.z;
    const int m0 = blockIdx.y * MG;
    int cnte = cnt[e]; if (cnte > CAPACITY) cnte = CAPACITY;
    if (m0 >= cnte) return;

    const int wave  = threadIdx.x >> 5;
    const int lane  = threadIdx.x & 31;
    const int nBase = blockIdx.x * 64 + wave * 16;
    const int halfSel = lane >> 4;
    const int mA      = lane & 15;
    const int kaBase  = halfSel * 16;
    const int kbBase  = halfSel * 16;
    const int nCol    = nBase + (lane & 15);

    __shared__ unsigned short aTile[MG][32];        // 64x32 bf16 activation panel
    const int r  = threadIdx.x >> 1;
    const int kq = (threadIdx.x & 1) * 16;
    const unsigned short* __restrict__ aSrc = act + (size_t)(e * CAPACITY + m0 + r) * I_DIM + kq;
    const float* __restrict__ wdE = wd + (size_t)e * I_DIM * H_DIM + nCol;

    v8f c[MT];
    #pragma unroll
    for (int mt = 0; mt < MT; ++mt) c[mt] = (v8f){};

    for (int ks = 0; ks < I_DIM; ks += 32) {
        // cooperative A panel copy (already bf16): async global->LDS when available
#if defined(USE_ASYNC_LDS)
        __builtin_amdgcn_global_load_async_to_lds_b128(
            (__attribute__((address_space(1))) v4i*)(aSrc + ks),
            (__attribute__((address_space(3))) v4i*)&aTile[r][kq], 0, 0);
        __builtin_amdgcn_global_load_async_to_lds_b128(
            (__attribute__((address_space(1))) v4i*)(aSrc + ks + 8),
            (__attribute__((address_space(3))) v4i*)&aTile[r][kq + 8], 0, 0);
        __builtin_amdgcn_s_wait_asynccnt(0);
#else
        const uint4* s128 = (const uint4*)(aSrc + ks);
        uint4* d128 = (uint4*)&aTile[r][kq];
        d128[0] = s128[0]; d128[1] = s128[1];
#endif
        __syncthreads();
        BFrag bfr;
        #pragma unroll
        for (int j = 0; j < 8; ++j) {
            const size_t koff = (size_t)(ks + kbBase + 2*j) * H_DIM;
            bfr.u[j] = pack2_bf16(wdE[koff], wdE[koff + H_DIM]);
        }
        #pragma unroll
        for (int mt = 0; mt < MT; ++mt) {
            BFrag af;
            const unsigned short* ar = &aTile[mt*16 + mA][0];
            #pragma unroll
            for (int j = 0; j < 4; ++j) {
                af.u[j]     = *(const unsigned int*)&ar[kaBase + 2*j];
                af.u[4 + j] = *(const unsigned int*)&ar[kaBase + 8 + 2*j];
            }
            c[mt] = __builtin_amdgcn_wmma_f32_16x16x32_bf16(false, af.v, false, bfr.v, (short)0, c[mt], false, false);
        }
        __syncthreads();
    }
    const int mTop = halfSel * 8;
    #pragma unroll
    for (int mt = 0; mt < MT; ++mt) {
        #pragma unroll
        for (int j = 0; j < 8; ++j) {
            const int gm = m0 + mt*16 + mTop + j;
            if (gm < cnte) {
                const int t  = slot_tok[e*CAPACITY + gm];
                const float w = slot_w[e*CAPACITY + gm];
                atomicAdd(out + (size_t)t * H_DIM + nCol, w * c[mt][j]);
            }
        }
    }
}

extern "C" void kernel_launch(void* const* d_in, const int* in_sizes, int n_in,
                              void* d_out, int out_size, void* d_ws, size_t ws_size,
                              hipStream_t stream) {
    (void)in_sizes; (void)n_in; (void)out_size; (void)ws_size;
    const float* hs   = (const float*)d_in[0];
    const float* rw   = (const float*)d_in[1];
    const float* bias = (const float*)d_in[2];
    const float* wg   = (const float*)d_in[3];
    const float* wu   = (const float*)d_in[4];
    const float* wd   = (const float*)d_in[5];
    float* out = (float*)d_out;

    char* ws = (char*)d_ws;
    int*   top_idx   = (int*)ws;    ws += (size_t)T_TOK * TOPK * 4;
    float* top_w     = (float*)ws;  ws += (size_t)T_TOK * TOPK * 4;
    float* zero_coef = (float*)ws;  ws += (size_t)T_TOK * 4;
    int*   cnt       = (int*)ws;    ws += (size_t)E_RTD * 4;
    int*   slot_tok  = (int*)ws;    ws += (size_t)E_RTD * CAPACITY * 4;
    float* slot_w    = (float*)ws;  ws += (size_t)E_RTD * CAPACITY * 4;
    uintptr_t p = ((uintptr_t)ws + 255) & ~(uintptr_t)255;
    unsigned short* act = (unsigned short*)p;   // [E, CAP, I] bf16 (~20 MB)

    router_kernel  <<<T_TOK, 128, 0, stream>>>(hs, rw, bias, top_idx, top_w, zero_coef);
    dispatch_kernel<<<1, 64, 0, stream>>>(top_idx, top_w, cnt, slot_tok, slot_w);
    out_init_kernel<<<(T_TOK * H_DIM / 4) / 256, 256, 0, stream>>>(hs, zero_coef, out);
    gateup_kernel  <<<dim3(I_DIM / 64, CAPACITY / MG, E_RTD), 128, 0, stream>>>(hs, wg, wu, cnt, slot_tok, act);
    down_kernel    <<<dim3(H_DIM / 64, CAPACITY / MG, E_RTD), 128, 0, stream>>>(act, wd, cnt, slot_tok, slot_w, out);
}
